// ElementwiseSparsity_75256416960823
// MI455X (gfx1250) — matrix operand order repaired and checked by reading, not compiled, not run on hardware
//
#include <hip/hip_runtime.h>
#include <hip/hip_bf16.h>

// ---------------------------------------------------------------------------
// Problem constants (fixed by the reference)
// ---------------------------------------------------------------------------
#define BATCH 8
#define CCH   128     // C
#define TT    4096    // T
#define HIGH  2048
#define KEEP  64

#define BM 128        // M tile (HIGH dim)
#define BN 128        // N tile (T dim)
#define XS_STRIDE 132 // padded LDS stride for transposed X tile (keeps dword align)

#define CAND_MAX 4096

// workspace layout (bytes)
#define GHIST_OFF    0            // 8*1024*4 = 32768
#define CCOUNT_OFF   32768        // 8*4
#define THRESH_OFF   32800        // 8*4
#define CAND_IDX_OFF 32832        // 8*4096*4
#define CAND_VAL_OFF 163904       // 8*4096*4
#define KEPT_IDX_OFF 294976       // 8*64*4
#define KEPT_VAL_OFF 297024       // 8*64*4
#define ZERO_BYTES   32800        // ghist + ccount must start at 0 each launch

typedef __attribute__((ext_vector_type(16))) __bf16 v16bf;
typedef __attribute__((ext_vector_type(8)))  float  v8f;

union ABFrag { v16bf v; unsigned u[8]; };

__device__ __forceinline__ unsigned short f2bf(float f) {
  unsigned u = __float_as_uint(f);
  u += 0x7fffu + ((u >> 16) & 1u);          // round-to-nearest-even
  return (unsigned short)(u >> 16);
}

// map float to monotonically ordered unsigned
__device__ __forceinline__ unsigned ordf(float f) {
  unsigned u = __float_as_uint(f);
  return (u & 0x80000000u) ? ~u : (u | 0x80000000u);
}

// ---------------------------------------------------------------------------
// Expand GEMM (bf16 WMMA, fp32 accumulate) with two epilogues:
//   MODE==0 : per-batch 1024-bin histogram of order-mapped h values
//   MODE==1 : collect candidates with ord(value) >= thresh[b]
// grid: (T/BN=32, HIGH/BM=16, BATCH), block: 256 threads = 8 waves
// ---------------------------------------------------------------------------
template <int MODE>
__global__ __launch_bounds__(256) void expand_gemm_kernel(
    const float* __restrict__ x,        // [B, C, T]
    const float* __restrict__ ew,       // [HIGH, C]
    const float* __restrict__ eb,       // [HIGH]
    int*   __restrict__ ghist,          // [B, 1024]   (MODE 0)
    const unsigned* __restrict__ thresh,// [B]         (MODE 1)
    int*   __restrict__ ccount,         // [B]         (MODE 1)
    int*   __restrict__ cidx,           // [B, CAND_MAX]
    float* __restrict__ cval)           // [B, CAND_MAX]
{
  __shared__ unsigned short Ws[BM * CCH];        // W tile, [m][k], bf16 bits
  __shared__ unsigned short Xs[BN * XS_STRIDE];  // X tile, [n][k] transposed
  __shared__ int hist[1024];

  const int tid = threadIdx.x;
  const int t0  = blockIdx.x * BN;
  const int m0  = blockIdx.y * BM;
  const int b   = blockIdx.z;

  if (MODE == 0) {
    for (int i = tid; i < 1024; i += 256) hist[i] = 0;
  }

  // ---- stage W tile: row r = tid/2, half = tid&1 covers 64 floats ----
  {
    const int r = tid >> 1, half = tid & 1;
    const float4* src = (const float4*)(ew + (size_t)(m0 + r) * CCH + half * 64);
    unsigned* dst = (unsigned*)&Ws[r * CCH + half * 64];
#pragma unroll
    for (int i = 0; i < 16; ++i) {
      float4 f = src[i];
      dst[i * 2 + 0] = (unsigned)f2bf(f.x) | ((unsigned)f2bf(f.y) << 16);
      dst[i * 2 + 1] = (unsigned)f2bf(f.z) | ((unsigned)f2bf(f.w) << 16);
    }
  }
  // ---- stage X tile transposed: k = tid/2, half = tid&1 covers 64 t ----
  {
    const int k = tid >> 1, half = tid & 1;
    const float4* src = (const float4*)(x + ((size_t)(b * CCH + k)) * TT + t0 + half * 64);
#pragma unroll
    for (int i = 0; i < 16; ++i) {
      float4 f = src[i];
      int n = half * 64 + i * 4;
      Xs[(n + 0) * XS_STRIDE + k] = f2bf(f.x);
      Xs[(n + 1) * XS_STRIDE + k] = f2bf(f.y);
      Xs[(n + 2) * XS_STRIDE + k] = f2bf(f.z);
      Xs[(n + 3) * XS_STRIDE + k] = f2bf(f.w);
    }
  }
  __syncthreads();

  // ---- wave tiling: wave computes 32(M) x 64(N) = 2x4 subtiles of 16x16 ----
  const int lane = tid & 31;
  const int wv   = tid >> 5;
  const int hi   = lane >> 4;     // lane half
  const int l16  = lane & 15;
  const int wm   = (wv & 3) * 32; // M offset within tile
  const int wn   = (wv >> 2) * 64;// N offset within tile

  v8f acc[2][4];
#pragma unroll
  for (int ms = 0; ms < 2; ++ms)
#pragma unroll
    for (int ns = 0; ns < 4; ++ns) acc[ms][ns] = {};

#pragma unroll
  for (int kk = 0; kk < 4; ++kk) {
    const int k0 = kk * 32;
    ABFrag a[2], bfr[4];
#pragma unroll
    for (int ms = 0; ms < 2; ++ms) {
      const int m = wm + ms * 16 + l16;
#pragma unroll
      for (int p = 0; p < 8; ++p) {
        const int kA = k0 + ((p >> 2) << 4) + (hi << 3) + ((p & 3) << 1);
        a[ms].u[p] = *(const unsigned*)&Ws[m * CCH + kA];
      }
    }
#pragma unroll
    for (int ns = 0; ns < 4; ++ns) {
      const int n = wn + ns * 16 + l16;
      const int base = n * XS_STRIDE + k0;
#pragma unroll
      for (int p = 0; p < 8; ++p)
        bfr[ns].u[p] = *(const unsigned*)&Xs[base + ((p >> 2) << 4) + (hi << 3) + ((p & 3) << 1)];
    }
#pragma unroll
    for (int ms = 0; ms < 2; ++ms)
#pragma unroll
      for (int ns = 0; ns < 4; ++ns)
        acc[ms][ns] = __builtin_amdgcn_wmma_f32_16x16x32_bf16(
            false, a[ms].v, false, bfr[ns].v, (short)0, acc[ms][ns],
            false, false);
  }

  // ---- epilogue: add bias, histogram or collect ----
  unsigned th = 0;
  if (MODE == 1) th = thresh[b];

#pragma unroll
  for (int ms = 0; ms < 2; ++ms) {
#pragma unroll
    for (int v = 0; v < 8; ++v) {
      const int o  = m0 + wm + ms * 16 + hi * 8 + v;
      const float bias = eb[o];
#pragma unroll
      for (int ns = 0; ns < 4; ++ns) {
        const int t = t0 + wn + ns * 16 + l16;
        const float val = acc[ms][ns][v] + bias;
        const unsigned u = ordf(val);
        if (MODE == 0) {
          atomicAdd(&hist[u >> 22], 1);
        } else {
          if (u >= th) {
            int pos = atomicAdd(&ccount[b], 1);
            if (pos < CAND_MAX) {
              cidx[b * CAND_MAX + pos] = o * TT + t;
              cval[b * CAND_MAX + pos] = val;
            }
          }
        }
      }
    }
  }

  if (MODE == 0) {
    __syncthreads();
    for (int i = tid; i < 1024; i += 256) {
      int c = hist[i];
      if (c) atomicAdd(&ghist[b * 1024 + i], c);
    }
  }
}

// ---------------------------------------------------------------------------
// Per-batch threshold: largest bin so that count(values in bins >= it) >= KEEP
// ---------------------------------------------------------------------------
__global__ void threshold_kernel(const int* __restrict__ ghist,
                                 unsigned* __restrict__ thresh) {
  int b = threadIdx.x;
  if (b >= BATCH) return;
  int cum = 0;
  unsigned th = 0;
  for (int bin = 1023; bin >= 0; --bin) {
    cum += ghist[b * 1024 + bin];
    if (cum >= KEEP) { th = ((unsigned)bin) << 22; break; }
  }
  thresh[b] = th;
}

// ---------------------------------------------------------------------------
// Exact top-KEEP among candidates (tie-break by flat index => deterministic
// regardless of atomic append order), then recompute kept values in full fp32.
// One block per batch.
// ---------------------------------------------------------------------------
__global__ __launch_bounds__(256) void select_kernel(
    const float* __restrict__ x, const float* __restrict__ ew,
    const float* __restrict__ eb,
    const int* __restrict__ ccount,
    const int* __restrict__ cidx, const float* __restrict__ cval,
    int* __restrict__ kept_idx, float* __restrict__ kept_val)
{
  __shared__ float cv[CAND_MAX];
  __shared__ int   ci[CAND_MAX];
  __shared__ float rv[256];
  __shared__ int   rp[256];
  __shared__ int   sel[KEEP];

  const int b = blockIdx.x;
  const int tid = threadIdx.x;
  int n = ccount[b];
  if (n > CAND_MAX) n = CAND_MAX;

  for (int i = tid; i < n; i += 256) { cv[i] = cval[b * CAND_MAX + i]; ci[i] = cidx[b * CAND_MAX + i]; }
  __syncthreads();

  for (int j = 0; j < KEEP; ++j) {
    float best = -3.0e38f; int bpos = -1; int bidx = 0x7fffffff;
    for (int i = tid; i < n; i += 256) {
      float v = cv[i]; int idx = ci[i];
      if (v > best || (v == best && idx < bidx)) { best = v; bpos = i; bidx = idx; }
    }
    rv[tid] = best; rp[tid] = bpos;
    __syncthreads();
    for (int s = 128; s > 0; s >>= 1) {
      if (tid < s) {
        float va = rv[tid], vb = rv[tid + s];
        int pa = rp[tid], pb = rp[tid + s];
        int ia = (pa >= 0) ? ci[pa] : 0x7fffffff;
        int ib = (pb >= 0) ? ci[pb] : 0x7fffffff;
        if (pb >= 0 && (pa < 0 || vb > va || (vb == va && ib < ia))) { rv[tid] = vb; rp[tid] = pb; }
      }
      __syncthreads();
    }
    if (tid == 0) {
      int p = rp[0];
      sel[j] = p;
      if (p >= 0) cv[p] = -3.0e38f;   // remove winner
    }
    __syncthreads();
  }

  // recompute exact fp32 values for the kept entries
  if (tid < KEEP) {
    int p = sel[tid];
    int idx = 0; float val = 0.0f;
    if (p >= 0) {
      idx = ci[p];
      const int o = idx / TT, t = idx % TT;
      float accv = eb[o];
      const float* wr = ew + (size_t)o * CCH;
      const float* xc = x + (size_t)b * CCH * TT + t;
      for (int c = 0; c < CCH; ++c) accv += wr[c] * xc[(size_t)c * TT];
      val = accv;
    }
    kept_idx[b * KEEP + tid] = idx;
    kept_val[b * KEEP + tid] = val;
  }
}

// ---------------------------------------------------------------------------
// Output: out[b,c,t] = contract_b[c] + sum over kept entries with t_j == t of
// contract_w[c, o_j] * v_j.   grid: (T/32, B), block 256 (covers 128c x 32t)
// ---------------------------------------------------------------------------
__global__ __launch_bounds__(256) void output_kernel(
    const float* __restrict__ cw,       // [C, HIGH]
    const float* __restrict__ cb,       // [C]
    const int* __restrict__ kept_idx, const float* __restrict__ kept_val,
    float* __restrict__ out)            // [B, C, T]
{
  __shared__ int   s_t[KEEP];
  __shared__ int   s_o[KEEP];
  __shared__ float s_v[KEEP];

  const int tid = threadIdx.x;
  const int t0  = blockIdx.x * 32;
  const int b   = blockIdx.y;

  if (tid < KEEP) {
    int idx = kept_idx[b * KEEP + tid];
    s_o[tid] = idx / TT;
    s_t[tid] = idx % TT;
    s_v[tid] = kept_val[b * KEEP + tid];
  }
  __syncthreads();

#pragma unroll
  for (int r = 0; r < 16; ++r) {
    const int lin = tid + r * 256;        // over 128c x 32t
    const int c = lin >> 5;
    const int t = t0 + (lin & 31);
    float acc = cb[c];
    for (int j = 0; j < KEEP; ++j)
      if (s_t[j] == t) acc += cw[(size_t)c * HIGH + s_o[j]] * s_v[j];
    out[((size_t)(b * CCH + c)) * TT + t] = acc;
  }
}

// ---------------------------------------------------------------------------
extern "C" void kernel_launch(void* const* d_in, const int* in_sizes, int n_in,
                              void* d_out, int out_size, void* d_ws, size_t ws_size,
                              hipStream_t stream) {
  const float* x  = (const float*)d_in[0];
  const float* ew = (const float*)d_in[1];
  const float* eb = (const float*)d_in[2];
  const float* cw = (const float*)d_in[3];
  const float* cb = (const float*)d_in[4];
  // d_in[5] = keep (fixed at 64)

  char* ws = (char*)d_ws;
  int*      ghist    = (int*)(ws + GHIST_OFF);
  int*      ccount   = (int*)(ws + CCOUNT_OFF);
  unsigned* thresh   = (unsigned*)(ws + THRESH_OFF);
  int*      cidx     = (int*)(ws + CAND_IDX_OFF);
  float*    cval     = (float*)(ws + CAND_VAL_OFF);
  int*      kept_idx = (int*)(ws + KEPT_IDX_OFF);
  float*    kept_val = (float*)(ws + KEPT_VAL_OFF);

  hipMemsetAsync(d_ws, 0, ZERO_BYTES, stream);   // zero histograms + counters

  dim3 gg(TT / BN, HIGH / BM, BATCH);

  // Pass 1: WMMA GEMM + histogram
  expand_gemm_kernel<0><<<gg, 256, 0, stream>>>(x, ew, eb, ghist,
                                                nullptr, nullptr, nullptr, nullptr);
  // Per-batch threshold from histogram
  threshold_kernel<<<1, BATCH, 0, stream>>>(ghist, thresh);
  // Pass 2: WMMA GEMM recompute + candidate collection
  expand_gemm_kernel<1><<<gg, 256, 0, stream>>>(x, ew, eb, nullptr,
                                                thresh, ccount, cidx, cval);
  // Exact top-64 per batch + fp32 value recompute
  select_kernel<<<BATCH, 256, 0, stream>>>(x, ew, eb, ccount, cidx, cval,
                                           kept_idx, kept_val);
  // Broadcast bias + sparse rank corrections
  output_kernel<<<dim3(TT / 32, BATCH), 256, 0, stream>>>(cw, cb, kept_idx,
                                                          kept_val, (float*)d_out);
}
